// SoftOptionCritic_32693291057940
// MI455X (gfx1250) — matrix operand order repaired and checked by reading, not compiled, not run on hardware
//
#include <hip/hip_runtime.h>
#include <math.h>

#define NB    8192
#define OBS   128
#define HID   512
#define HEADS 4
#define VSZ   256
#define NU    64
#define TOPK  8
#define NOPT  16

// fp32 h tile in LDS: row stride 520 floats -> bank stride 8 dwords/row,
// so the two b128 reads per lane in phase 2 are conflict-free.
#define H_STRIDE 520

typedef __attribute__((ext_vector_type(8)))  float  v8f;
typedef __attribute__((ext_vector_type(16))) __bf16 v16bf;

// ---------------------------------------------------------------------------
// bf16 error-compensated split of 16 f32 (two 32B-aligned groups of 8:
// K..K+7 and K+16..K+23) into WMMA A-fragment order.
// ---------------------------------------------------------------------------
__device__ __forceinline__ void split16(const float* __restrict__ p0,
                                        const float* __restrict__ p1,
                                        v16bf& hi, v16bf& lo) {
  v8f f0 = *(const v8f*)p0;   // 32B aligned
  v8f f1 = *(const v8f*)p1;
#pragma unroll
  for (int e = 0; e < 8; ++e) {
    float a = f0[e];
    __bf16 ah = (__bf16)a;
    hi[e] = ah;
    lo[e] = (__bf16)(a - (float)ah);
    float b = f1[e];
    __bf16 bh = (__bf16)b;
    hi[e + 8] = bh;
    lo[e + 8] = (__bf16)(b - (float)bh);
  }
}

__device__ __forceinline__ v8f wmma_bf16(v16bf a, v16bf b, v8f c) {
  return __builtin_amdgcn_wmma_f32_16x16x32_bf16(
      false, a, false, b, (short)0, c, false, false);
}

// ---------------------------------------------------------------------------
// Prep 1: scale[o] = (sum_{n in top8} sigmoid(p_w[o,n]+p_b[n]))/256
//         cvec[v]  = (sum_h value_b[h*256+v])/32
// ---------------------------------------------------------------------------
__global__ void __launch_bounds__(256) soc_prep_small(
    const float* __restrict__ p_w, const float* __restrict__ p_b,
    const float* __restrict__ value_b, float* __restrict__ scale,
    float* __restrict__ cvec) {
  const int t = threadIdx.x;
  if (t < VSZ) {
    cvec[t] = (value_b[t] + value_b[VSZ + t] + value_b[2 * VSZ + t] +
               value_b[3 * VSZ + t]) * (1.0f / 32.0f);
  }
  if (t < NOPT) {
    float s[NU];
    for (int n = 0; n < NU; ++n) s[n] = p_w[t * NU + n] + p_b[n];
    float acc = 0.0f;
    for (int k = 0; k < TOPK; ++k) {
      int best = 0;
      float bv = s[0];
      for (int n = 1; n < NU; ++n)
        if (s[n] > bv) { bv = s[n]; best = n; }   // ties -> lowest index (jax)
      acc += 1.0f / (1.0f + expf(-bv));           // softmax over {s,0} = sigmoid
      s[best] = -3.402823466e38f;
    }
    scale[t] = acc * (1.0f / 256.0f);
  }
}

// ---------------------------------------------------------------------------
// Pack weights into WMMA-B-native bf16 hi/lo chunks.
// Chunk layout: elem index = ((k0/32)*N + n)*32 + half*16 + e,
//   value = W[k0 + half*8 + (e&7) + (e>>3)*16][n]   (half = lane>>4 at load time)
// => each lane's B fragment is 32 contiguous bytes (2x global_load_b128).
// ---------------------------------------------------------------------------
__global__ void __launch_bounds__(256) soc_pack_w1(
    const float* __restrict__ w1, __bf16* __restrict__ hi,
    __bf16* __restrict__ lo) {
  const int i = blockIdx.x * 256 + threadIdx.x;   // OBS*HID = 65536
  if (i < OBS * HID) {
    const int e     = i & 15;
    const int chunk = i >> 4;
    const int half  = chunk & 1;
    const int n     = (chunk >> 1) & (HID - 1);
    const int k0    = ((chunk >> 1) / HID) << 5;
    const float v = w1[(k0 + half * 8 + (e & 7) + (e >> 3) * 16) * HID + n];
    const __bf16 h = (__bf16)v;
    hi[i] = h;
    lo[i] = (__bf16)(v - (float)h);
  }
}

// Same packing for W2'[k][v] = sum_h value_w[k, h*256+v]  (512x256)
__global__ void __launch_bounds__(256) soc_pack_w2(
    const float* __restrict__ value_w, __bf16* __restrict__ hi,
    __bf16* __restrict__ lo) {
  const int i = blockIdx.x * 256 + threadIdx.x;   // HID*VSZ = 131072
  if (i < HID * VSZ) {
    const int e     = i & 15;
    const int chunk = i >> 4;
    const int half  = chunk & 1;
    const int n     = (chunk >> 1) & (VSZ - 1);
    const int k0    = ((chunk >> 1) / VSZ) << 5;
    const float* r = value_w + (k0 + half * 8 + (e & 7) + (e >> 3) * 16) * (HEADS * VSZ);
    const float v = r[n] + r[VSZ + n] + r[2 * VSZ + n] + r[3 * VSZ + n];
    const __bf16 h = (__bf16)v;
    hi[i] = h;
    lo[i] = (__bf16)(v - (float)h);
  }
}

// ---------------------------------------------------------------------------
// Fused double-GEMM, 3x-bf16 split on v_wmma_f32_16x16x32_bf16:
//   h(32x512) = x(32x128) @ w1 + b1            (phase 1 -> LDS, fp32)
//   out(32x256) = (h @ w2')*scale[opt] + cvec  (phase 2)
// 256 threads = 8 waves (wave32); block owns 32 batch rows.
// ---------------------------------------------------------------------------
__global__ void __launch_bounds__(256) soc_fused(
    const float* __restrict__ x, const __bf16* __restrict__ w1p_hi,
    const __bf16* __restrict__ w1p_lo, const float* __restrict__ b1,
    const __bf16* __restrict__ w2p_hi, const __bf16* __restrict__ w2p_lo,
    const float* __restrict__ scale, const float* __restrict__ cvec,
    const int* __restrict__ option, float* __restrict__ out) {
  __shared__ float h_lds[32 * H_STRIDE];
  __shared__ float row_scale[32];

  const int lane = threadIdx.x & 31;
  const int wave = threadIdx.x >> 5;
  const int m    = lane & 15;          // A row / B,C,D column within tile
  const int half = lane >> 4;          // 16-bit WMMA K-half select
  const int koff = half * 8;           // A-frag K offset for this half
  const int row0 = blockIdx.x * 32;

  if (threadIdx.x < 32)
    row_scale[threadIdx.x] = scale[option[row0 + threadIdx.x]];

  const int rt    = wave & 1;          // row tile (0/1)
  const int cbase = (wave >> 1) * 16;  // base column tile for this wave
  const int arow  = rt * 16 + m;       // this lane's A row within block

  // ---- phase 1: h = x @ w1 + b1 (A from global x, B prepacked bf16) ----
  v8f acc[8] = {};
  const float* xrow = x + (row0 + arow) * OBS;
  const int bcol1 = (cbase + m) * 32 + half * 16;  // per-lane B chunk offset
  for (int k0 = 0; k0 < OBS; k0 += 32) {
    v16bf a_hi, a_lo;
    split16(xrow + k0 + koff, xrow + k0 + koff + 16, a_hi, a_lo);
    const __bf16* bh_base = w1p_hi + (k0 >> 5) * (HID * 32) + bcol1;
    const __bf16* bl_base = w1p_lo + (k0 >> 5) * (HID * 32) + bcol1;
#pragma unroll
    for (int t = 0; t < 8; ++t) {
      const v16bf b_hi = *(const v16bf*)(bh_base + t * 2048);
      const v16bf b_lo = *(const v16bf*)(bl_base + t * 2048);
      acc[t] = wmma_bf16(a_hi, b_hi, acc[t]);
      acc[t] = wmma_bf16(a_lo, b_hi, acc[t]);
      acc[t] = wmma_bf16(a_hi, b_lo, acc[t]);
    }
  }

  // write fp32 h tile (+bias) into LDS
#pragma unroll
  for (int t = 0; t < 8; ++t) {
    const int col = cbase + t * 64 + m;
    const float bias = b1[col];
#pragma unroll
    for (int j = 0; j < 8; ++j) {
      const int r = rt * 16 + j + half * 8;  // C layout: vgpr j -> M=j+8*half
      h_lds[r * H_STRIDE + col] = acc[t][j] + bias;
    }
  }
  __syncthreads();

  // ---- phase 2: out = h @ w2' (A split from LDS fp32, B prepacked) ----
  v8f acc2[4] = {};
  const float* hrow = h_lds + arow * H_STRIDE;
  const int bcol2 = (cbase + m) * 32 + half * 16;
  for (int k0 = 0; k0 < HID; k0 += 32) {
    v16bf a_hi, a_lo;
    split16(hrow + k0 + koff, hrow + k0 + koff + 16, a_hi, a_lo);
    const __bf16* bh_base = w2p_hi + (k0 >> 5) * (VSZ * 32) + bcol2;
    const __bf16* bl_base = w2p_lo + (k0 >> 5) * (VSZ * 32) + bcol2;
#pragma unroll
    for (int t = 0; t < 4; ++t) {
      const v16bf b_hi = *(const v16bf*)(bh_base + t * 2048);
      const v16bf b_lo = *(const v16bf*)(bl_base + t * 2048);
      acc2[t] = wmma_bf16(a_hi, b_hi, acc2[t]);
      acc2[t] = wmma_bf16(a_lo, b_hi, acc2[t]);
      acc2[t] = wmma_bf16(a_hi, b_lo, acc2[t]);
    }
  }

#pragma unroll
  for (int t = 0; t < 4; ++t) {
    const int col = cbase + t * 64 + m;
    const float cv = cvec[col];
#pragma unroll
    for (int j = 0; j < 8; ++j) {
      const int r = rt * 16 + j + half * 8;
      out[(row0 + r) * VSZ + col] = acc2[t][j] * row_scale[r] + cv;
    }
  }
}

// ---------------------------------------------------------------------------
extern "C" void kernel_launch(void* const* d_in, const int* in_sizes, int n_in,
                              void* d_out, int out_size, void* d_ws, size_t ws_size,
                              hipStream_t stream) {
  const float* x   = (const float*)d_in[0];   // (8192,128)
  const int*   opt = (const int*)d_in[1];     // (8192,)
  const float* w1  = (const float*)d_in[2];   // (128,512)
  const float* b1  = (const float*)d_in[3];   // (512,)
  const float* vw  = (const float*)d_in[4];   // (512,1024)
  const float* vb  = (const float*)d_in[5];   // (1024,)
  const float* pw  = (const float*)d_in[6];   // (16,64)
  const float* pb  = (const float*)d_in[7];   // (64,)
  float* out = (float*)d_out;                 // (8192,256)

  float* ws    = (float*)d_ws;
  float* scale = ws;          // 16 floats
  float* cvec  = ws + 16;     // 256 floats
  __bf16* w1p_hi = (__bf16*)((char*)d_ws + 4096);   // 65536 bf16
  __bf16* w1p_lo = w1p_hi + OBS * HID;              // 65536 bf16
  __bf16* w2p_hi = w1p_lo + OBS * HID;              // 131072 bf16
  __bf16* w2p_lo = w2p_hi + HID * VSZ;              // 131072 bf16  (~772KB total)

  soc_prep_small<<<1, 256, 0, stream>>>(pw, pb, vb, scale, cvec);
  soc_pack_w1<<<(OBS * HID) / 256, 256, 0, stream>>>(w1, w1p_hi, w1p_lo);
  soc_pack_w2<<<(HID * VSZ) / 256, 256, 0, stream>>>(vw, w2p_hi, w2p_lo);
  soc_fused<<<NB / 32, 256, 0, stream>>>(x, w1p_hi, w1p_lo, b1, w2p_hi, w2p_lo,
                                         scale, cvec, opt, out);
}
